// RNNLanguageModel_10599979287079
// MI455X (gfx1250) — matrix operand order repaired and checked by reading, not compiled, not run on hardware
//
#include <hip/hip_runtime.h>
#include <hip/hip_bf16.h>
#include <math.h>

// RNN LM: B=32 T=512 V=256 E=512 H=1024
// MI455X / gfx1250, wave32, v_wmma_f32_16x16x32_bf16, fp32 accumulation.
//  1) params -> bf16 (one pass)
//  2) X0 = gather(embed)[x] @ Wx0^T + b0       (parallel WMMA GEMM)
//  3) persistent 32-WG skewed scan, ONE grid barrier/step:
//       per step i: h0(i) = tanh(X0(i) + h0(i-1)@Wh0^T)
//                   h1(i-1) = tanh(h0(i-1)@Wx1^T + h1(i-2)@Wh1^T + b1)
//     both share A = h0(i-1): fused k-loop = 1 A-fragment -> 2 WMMAs.
//     Weights pinned in LDS (192KB/WGP). Global A-fragments use a DEPTH-4
//     rotating pipeline (prefetch k+128) so the WMMA wait has ~4 iters of
//     slack instead of serializing on L2 latency every iteration.
//  4) logits = Hout @ Wout^T + bout            (parallel WMMA GEMM)

#define Bq 32
#define Tq 512
#define Vq 256
#define Eq 512
#define Hq 1024
#define NWG 32

typedef __attribute__((ext_vector_type(16))) __bf16 bf16x16;
typedef __attribute__((ext_vector_type(8)))  float  f32x8;

union AB { bf16x16 v; unsigned u[8]; };

__device__ __forceinline__ unsigned short f2bf(float f) {
  unsigned u = __float_as_uint(f);
  u = (u + 0x7FFFu + ((u >> 16) & 1u)) >> 16;   // round-to-nearest-even
  return (unsigned short)u;
}

// A-matrix 16x32 bf16: lane L: m=L&15, half h=L>>4.
// VGPR j<4: K=8h+2j(+1); j>=4: K=16+8h+2(j-4)(+1)
__device__ __forceinline__ int a_klocal(int j, int h) {
  return (j < 4) ? (8 * h + 2 * j) : (16 + 8 * h + 2 * (j - 4));
}
// B-matrix 32x16 (B = W^T): lane n=L&15, half h: VGPR j<4: K=8h+2j(+1); j>=4: K=16+8h+2(j-4)(+1)
__device__ __forceinline__ int b_klocal(int j, int h) {
  return (j < 4) ? (8 * h + 2 * j) : (16 + 8 * h + 2 * (j - 4));
}

__device__ __forceinline__ void load_a(AB& a, const unsigned short* arow,
                                       int k0, int hh) {
#pragma unroll
  for (int j = 0; j < 8; ++j)
    a.u[j] = *(const unsigned*)(arow + k0 + a_klocal(j, hh));
}
__device__ __forceinline__ void load_b(AB& w, const unsigned short* brow,
                                       int k0, int hh) {
#pragma unroll
  for (int j = 0; j < 8; ++j)
    w.u[j] = *(const unsigned*)(brow + k0 + b_klocal(j, hh));
}

__device__ __forceinline__ f32x8 wmma_bf16(const AB& a, const AB& b, f32x8 c) {
  return __builtin_amdgcn_wmma_f32_16x16x32_bf16(false, a.v, false, b.v,
                                                 (short)0, c, false, false);
}

// ---------------- small utility kernels ----------------
__global__ void k_cvt_bf16(const float* __restrict__ src,
                           unsigned short* __restrict__ dst, int n) {
  int i = blockIdx.x * blockDim.x + threadIdx.x;
  if (i < n) dst[i] = f2bf(src[i]);
}

__global__ void k_zero_u32(unsigned* __restrict__ p, int n) {
  int i = blockIdx.x * blockDim.x + threadIdx.x;
  if (i < n) p[i] = 0u;
}

// ---------------- X0 = emb[x] @ Wx0^T + b0 ----------------
// grid (B*T/16, H/512), block 256 (8 waves); wave = 16 rows x 64 cols
__global__ void k_x0(const int* __restrict__ tok,
                     const unsigned short* __restrict__ embbf,
                     const unsigned short* __restrict__ wx0,
                     const float* __restrict__ b0,
                     float* __restrict__ X0) {
  const int lane = threadIdx.x & 31;
  const int wave = threadIdx.x >> 5;
  const int row0 = blockIdx.x * 16;
  const int col0 = blockIdx.y * 512 + wave * 64;
  const int m = lane & 15, hh = lane >> 4, n = lane & 15;

  const int tk = tok[row0 + m];                 // fused embedding gather
  const unsigned short* arow = embbf + tk * Eq;

  f32x8 c[4];
#pragma unroll
  for (int i = 0; i < 4; ++i) {
    float bias = b0[col0 + i * 16 + n];
#pragma unroll
    for (int r = 0; r < 8; ++r) c[i][r] = bias;
  }

  AB a;
  load_a(a, arow, 0, hh);
#pragma unroll 2
  for (int k0 = 0; k0 < Eq; k0 += 32) {
    AB an;
    load_a(an, arow, (k0 + 32) & (Eq - 1), hh);   // wrap prefetch
#pragma unroll
    for (int i = 0; i < 4; ++i) {
      AB b;
      load_b(b, wx0 + (col0 + i * 16 + n) * Eq, k0, hh);
      c[i] = wmma_bf16(a, b, c[i]);
    }
    a = an;
  }
#pragma unroll
  for (int i = 0; i < 4; ++i)
#pragma unroll
    for (int r = 0; r < 8; ++r) {
      int row = row0 + hh * 8 + r;
      X0[row * Hq + col0 + i * 16 + n] = c[i][r];
    }
}

// ---------------- persistent scan ----------------
__device__ __forceinline__ void grid_barrier(unsigned* bar, unsigned& gen) {
  __syncthreads();
  if (threadIdx.x == 0) {
    __threadfence();  // make this WG's global stores device-visible
    unsigned prev = __hip_atomic_fetch_add(&bar[0], 1u, __ATOMIC_ACQ_REL,
                                           __HIP_MEMORY_SCOPE_AGENT);
    unsigned target = gen + 1;
    if (prev == NWG - 1) {
      __hip_atomic_store(&bar[0], 0u, __ATOMIC_RELAXED, __HIP_MEMORY_SCOPE_AGENT);
      __hip_atomic_store(&bar[1], target, __ATOMIC_RELEASE, __HIP_MEMORY_SCOPE_AGENT);
    } else {
      while (__hip_atomic_load(&bar[1], __ATOMIC_ACQUIRE,
                               __HIP_MEMORY_SCOPE_AGENT) != target) {
        // plain spin: poll period ~= L2 latency already
      }
    }
  }
  __syncthreads();
  gen += 1;
}

// grid = NWG(32) x 128 threads (4 waves). WG owns 32 output columns; weight
// slices live in LDS. Skewed pipeline: one grid barrier per step.
// State convention: h0(s) in h0buf[s&1], h1(s) in h1buf[s&1]; initial states
// h0(-1), h1(-1) are zeros in buffer parity 1.
__global__ void k_scan(const float* __restrict__ X0,
                       const float* __restrict__ b1,
                       const unsigned short* __restrict__ Wh0,
                       const unsigned short* __restrict__ Wx1,
                       const unsigned short* __restrict__ Wh1,
                       unsigned short* __restrict__ h0buf,  // [2][B*H]
                       unsigned short* __restrict__ h1buf,  // [2][B*H]
                       unsigned short* __restrict__ Hout,   // [B*T][H] bf16
                       unsigned* __restrict__ bar) {
  extern __shared__ char smem_raw[];
  unsigned short* sWh0 = (unsigned short*)smem_raw;   // [32][H]
  unsigned short* sWx1 = sWh0 + 32 * Hq;
  unsigned short* sWh1 = sWx1 + 32 * Hq;

  const int tid = threadIdx.x;
  const int lane = tid & 31;
  const int wave = tid >> 5;
  const int colwg = blockIdx.x * 32;

  // stage weight slices (contiguous rows colwg..colwg+31), dword copies
  {
    const unsigned* s0 = (const unsigned*)(Wh0 + colwg * Hq);
    const unsigned* s1 = (const unsigned*)(Wx1 + colwg * Hq);
    const unsigned* s2 = (const unsigned*)(Wh1 + colwg * Hq);
    unsigned* d0 = (unsigned*)sWh0;
    unsigned* d1 = (unsigned*)sWx1;
    unsigned* d2 = (unsigned*)sWh1;
    for (int i = tid; i < 32 * Hq / 2; i += blockDim.x) {
      d0[i] = s0[i]; d1[i] = s1[i]; d2[i] = s2[i];
    }
  }
  __syncthreads();

  const int mt = wave >> 1;                 // row tile 0..1
  const int nt = wave & 1;                  // col tile 0..1
  const int m = lane & 15, hh = lane >> 4, n = lane & 15;
  const int rowb = mt * 16 + m;             // batch row for A fragments
  const int col = colwg + nt * 16 + n;      // global output column
  const unsigned short* sB0 = sWh0 + (nt * 16 + n) * Hq;
  const unsigned short* sB1 = sWx1 + (nt * 16 + n) * Hq;
  const unsigned short* sB2 = sWh1 + (nt * 16 + n) * Hq;
  const float bias1 = b1[col];

  unsigned gen = 0;
  for (int i = 0; i <= Tq; ++i) {
    const unsigned short* h0prev  = h0buf + ((i ^ 1) & 1) * (Bq * Hq); // h0(i-1)
    unsigned short*       h0next  = h0buf + (i & 1)       * (Bq * Hq); // h0(i)
    const unsigned short* h1prev2 = h1buf + (i & 1)       * (Bq * Hq); // h1(i-2)
    unsigned short*       h1next  = h1buf + ((i ^ 1) & 1) * (Bq * Hq); // h1(i-1)

    // warm next step's X0 rows (global_prefetch_b8)
    if (i + 1 < Tq) {
#pragma unroll
      for (int r = 0; r < 8; ++r) {
        int b = mt * 16 + hh * 8 + r;
        __builtin_prefetch(&X0[(b * Tq + (i + 1)) * Hq + col], 0, 1);
      }
    }

    // ---- fused k-loop over A = h0(i-1): c0 += A@Wh0^T, c1 += A@Wx1^T ----
    const int ti = (i < Tq) ? i : 0;        // clamp (c0 unused at i==Tq)
    f32x8 c0, c1;
#pragma unroll
    for (int r = 0; r < 8; ++r) {
      int b = mt * 16 + hh * 8 + r;
      c0[r] = X0[(b * Tq + ti) * Hq + col];
      c1[r] = bias1;
    }
    {
      const unsigned short* arow = h0prev + rowb * Hq;
      AB a[4];                               // depth-4 global pipeline
#pragma unroll
      for (int p = 0; p < 4; ++p) load_a(a[p], arow, 32 * p, hh);
      AB w0, w1;                             // depth-1 LDS pipeline
      load_b(w0, sB0, 0, hh);
      load_b(w1, sB1, 0, hh);
#pragma unroll 4
      for (int k0 = 0; k0 < Hq; k0 += 32) {
        const int idx = (k0 >> 5) & 3;
        const int kn = (k0 + 32)  & (Hq - 1);
        const int ka = (k0 + 128) & (Hq - 1); // wrap prefetch, 4 iters ahead
        AB an, w0n, w1n;
        load_a(an, arow, ka, hh);
        load_b(w0n, sB0, kn, hh);
        load_b(w1n, sB1, kn, hh);
        c0 = wmma_bf16(a[idx], w0, c0);
        c1 = wmma_bf16(a[idx], w1, c1);
        a[idx] = an;
        w0 = w0n; w1 = w1n;
      }
    }
    if (i < Tq) {
#pragma unroll
      for (int r = 0; r < 8; ++r) {
        int b = mt * 16 + hh * 8 + r;
        h0next[b * Hq + col] = f2bf(tanhf(c0[r]));   // h0(i)
      }
    }

    // ---- second k-loop over A = h1(i-2): c1 += A@Wh1^T; emit h1(i-1) ----
    if (i > 0) {
      const unsigned short* arow = h1prev2 + rowb * Hq;
      AB a[4];
#pragma unroll
      for (int p = 0; p < 4; ++p) load_a(a[p], arow, 32 * p, hh);
      AB w2;
      load_b(w2, sB2, 0, hh);
#pragma unroll 4
      for (int k0 = 0; k0 < Hq; k0 += 32) {
        const int idx = (k0 >> 5) & 3;
        const int kn = (k0 + 32)  & (Hq - 1);
        const int ka = (k0 + 128) & (Hq - 1);
        AB an, w2n;
        load_a(an, arow, ka, hh);
        load_b(w2n, sB2, kn, hh);
        c1 = wmma_bf16(a[idx], w2, c1);
        a[idx] = an;
        w2 = w2n;
      }
#pragma unroll
      for (int r = 0; r < 8; ++r) {
        int b = mt * 16 + hh * 8 + r;
        unsigned short hv = f2bf(tanhf(c1[r]));
        h1next[b * Hq + col] = hv;                       // h1(i-1)
        Hout[(b * Tq + (i - 1)) * Hq + col] = hv;
      }
    }
    grid_barrier(bar, gen);
  }
}

// ---------------- logits = Hout @ Wout^T + bout ----------------
// grid B*T/16, block 256 (8 waves); wave = 16 rows x 32 cols
__global__ void k_logits(const unsigned short* __restrict__ Hout,
                         const unsigned short* __restrict__ wout,
                         const float* __restrict__ bout,
                         float* __restrict__ out) {
  const int lane = threadIdx.x & 31;
  const int wave = threadIdx.x >> 5;
  const int row0 = blockIdx.x * 16;
  const int col0 = wave * 32;
  const int m = lane & 15, hh = lane >> 4, n = lane & 15;

  f32x8 c[2];
#pragma unroll
  for (int i = 0; i < 2; ++i) {
    float bias = bout[col0 + i * 16 + n];
#pragma unroll
    for (int r = 0; r < 8; ++r) c[i][r] = bias;
  }
  const unsigned short* arow = Hout + (row0 + m) * Hq;
  const unsigned short* brow0 = wout + (col0 + n) * Hq;
  const unsigned short* brow1 = wout + (col0 + 16 + n) * Hq;

  AB a, b0f, b1f;
  load_a(a, arow, 0, hh);
  load_b(b0f, brow0, 0, hh);
  load_b(b1f, brow1, 0, hh);
#pragma unroll 2
  for (int k0 = 0; k0 < Hq; k0 += 32) {
    const int kn = (k0 + 32) & (Hq - 1);
    AB an, b0n, b1n;
    load_a(an, arow, kn, hh);
    load_b(b0n, brow0, kn, hh);
    load_b(b1n, brow1, kn, hh);
    c[0] = wmma_bf16(a, b0f, c[0]);
    c[1] = wmma_bf16(a, b1f, c[1]);
    a = an; b0f = b0n; b1f = b1n;
  }
#pragma unroll
  for (int i = 0; i < 2; ++i)
#pragma unroll
    for (int r = 0; r < 8; ++r) {
      int row = row0 + hh * 8 + r;
      out[row * Vq + col0 + i * 16 + n] = c[i][r];
    }
}

// ---------------- host launch ----------------
extern "C" void kernel_launch(void* const* d_in, const int* in_sizes, int n_in,
                              void* d_out, int out_size, void* d_ws, size_t ws_size,
                              hipStream_t stream) {
  const int*   x     = (const int*)  d_in[0];
  const float* embed = (const float*)d_in[1];
  const float* Wx0   = (const float*)d_in[2];
  const float* Wh0   = (const float*)d_in[3];
  const float* b0    = (const float*)d_in[4];
  const float* Wx1   = (const float*)d_in[5];
  const float* Wh1   = (const float*)d_in[6];
  const float* b1    = (const float*)d_in[7];
  const float* Wout  = (const float*)d_in[8];
  const float* bout  = (const float*)d_in[9];

  char* ws = (char*)d_ws;
  size_t off = 0;
  auto alloc = [&](size_t bytes) -> char* {
    char* p = ws + off;
    off += (bytes + 255) & ~(size_t)255;
    return p;
  };
  unsigned*       bar    = (unsigned*)      alloc(256);
  unsigned short* embbf  = (unsigned short*)alloc((size_t)Vq * Eq * 2);
  unsigned short* wx0bf  = (unsigned short*)alloc((size_t)Hq * Eq * 2);
  unsigned short* wh0bf  = (unsigned short*)alloc((size_t)Hq * Hq * 2);
  unsigned short* wx1bf  = (unsigned short*)alloc((size_t)Hq * Hq * 2);
  unsigned short* wh1bf  = (unsigned short*)alloc((size_t)Hq * Hq * 2);
  unsigned short* woutbf = (unsigned short*)alloc((size_t)Vq * Hq * 2);
  unsigned short* h0buf  = (unsigned short*)alloc((size_t)2 * Bq * Hq * 2);
  unsigned short* h1buf  = (unsigned short*)alloc((size_t)2 * Bq * Hq * 2);
  unsigned short* Hout   = (unsigned short*)alloc((size_t)Bq * Tq * Hq * 2);
  float*          X0     = (float*)         alloc((size_t)Bq * Tq * Hq * 4);

  auto cvt = [&](const float* s, unsigned short* d, int n) {
    k_cvt_bf16<<<(n + 255) / 256, 256, 0, stream>>>(s, d, n);
  };
  cvt(embed, embbf, Vq * Eq);
  cvt(Wx0,   wx0bf, Hq * Eq);
  cvt(Wh0,   wh0bf, Hq * Hq);
  cvt(Wx1,   wx1bf, Hq * Hq);
  cvt(Wh1,   wh1bf, Hq * Hq);
  cvt(Wout,  woutbf, Vq * Hq);

  k_zero_u32<<<1, 64, 0, stream>>>(bar, 2);
  // zero BOTH parities of the state double buffers (initial states at parity 1)
  k_zero_u32<<<(Bq * Hq + 255) / 256, 256, 0, stream>>>((unsigned*)h0buf, Bq * Hq);
  k_zero_u32<<<(Bq * Hq + 255) / 256, 256, 0, stream>>>((unsigned*)h1buf, Bq * Hq);

  k_x0<<<dim3(Bq * Tq / 16, Hq / 512), 256, 0, stream>>>(x, embbf, wx0bf, b0, X0);

  size_t lds = (size_t)3 * 32 * Hq * sizeof(unsigned short);  // 192 KB
  (void)hipFuncSetAttribute(reinterpret_cast<const void*>(k_scan),
                            hipFuncAttributeMaxDynamicSharedMemorySize, (int)lds);
  k_scan<<<NWG, 128, lds, stream>>>(X0, b1, wh0bf, wx1bf, wh1bf,
                                    h0buf, h1buf, Hout, bar);

  k_logits<<<Bq * Tq / 16, 256, 0, stream>>>(Hout, woutbf, bout, (float*)d_out);
}
